// HierarchicalAttentionFusion_12429635355016
// MI455X (gfx1250) — matrix operand order repaired and checked by reading, not compiled, not run on hardware
//
#include <hip/hip_runtime.h>

// ---------------------------------------------------------------------------
// MI455X (gfx1250) implementation notes:
//  * wave32, WMMA 16x16x32 bf16 (fp32 accum) for all big GEMMs + attention.
//  * Whole working set (~90MB) is L2-resident (192MB); compute-bound, so the
//    win is moving the ~85 GFLOP of matrix math onto the WMMA pipes in bf16.
//  * Fragment layouts (A: per-lane row, contiguous K = 16*(l>>4)+s, not the ISA
//    table order; B: per-lane column with the same contiguous K; C/D: m=r+8*(l>>4),
//    n=l&15).
// ---------------------------------------------------------------------------

typedef __bf16 bf16;
typedef __bf16 v16bf __attribute__((ext_vector_type(16)));
typedef float  v8f   __attribute__((ext_vector_type(8)));
typedef unsigned int u32x4 __attribute__((ext_vector_type(4)));

union Frag { v16bf v; u32x4 q[2]; };

#define KTOP 614
#define TOK  2049
#define TPAD 2080   // keys padded to multiple of 32 for flash tiles

// activation codes
#define ACT_NONE 0
#define ACT_RELU 1
#define ACT_GELU 2
#define ACT_TANH 3
#define ACT_SIGM 4

__device__ __forceinline__ float apply_act(float x, int act) {
    switch (act) {
        case ACT_RELU: return fmaxf(x, 0.f);
        case ACT_GELU: return 0.5f * x * (1.f + erff(x * 0.70710678118654752f));
        case ACT_TANH: return tanhf(x);
        case ACT_SIGM: return 1.f / (1.f + expf(-x));
        default:       return x;
    }
}

__device__ __forceinline__ v8f zero8() {
    v8f z = {0.f,0.f,0.f,0.f,0.f,0.f,0.f,0.f};
    return z;
}

// ---------------------------------------------------------------------------
// Generic WMMA GEMM:  out[M,N] = act( A[M,K] * W[N,K]^T + bias + extra + resid )
// A, W bf16 row-major; fp32 accumulate; optional fp32 and/or bf16 outputs.
// Block = 256 threads = 8 waves arranged 4(M) x 2(N); tile 64x64.
// ---------------------------------------------------------------------------
__global__ __launch_bounds__(256)
void gemm_wmma_kernel(const bf16* __restrict__ A, int lda,
                      const bf16* __restrict__ W, int ldw,
                      const float* __restrict__ bias,
                      const float* __restrict__ extrab, int extraRows,
                      const float* resid,
                      float* outF, bf16* outB, int ldc,
                      int M, int N, int K, int act)
{
    const int wid  = threadIdx.x >> 5;
    const int lane = threadIdx.x & 31;
    const int wm = wid & 3, wn = wid >> 2;
    const int hh = lane >> 4, nl = lane & 15;

    const int row0 = blockIdx.x * 64 + wm * 16;
    const int col0 = blockIdx.y * 64 + wn * 32;

    int arow = row0 + nl; if (arow >= M) arow = M - 1;     // clamp; stores guarded
    const bf16* ap  = A + (long long)arow * lda + hh * 16;
    const bf16* bp0 = W + (long long)(col0 + nl)      * ldw + hh * 16;
    const bf16* bp1 = W + (long long)(col0 + 16 + nl) * ldw + hh * 16;

    v8f acc0 = zero8(), acc1 = zero8();

    for (int k0 = 0; k0 < K; k0 += 32) {
        __builtin_prefetch((const void*)(ap  + k0 + 128), 0, 1);  // global_prefetch_b8
        __builtin_prefetch((const void*)(bp0 + k0 + 128), 0, 1);
        Frag af, bf0, bf1;
        af.q[0]  = *(const u32x4*)(ap + k0);            // K = k0 + hh*16 + 0..7
        af.q[1]  = *(const u32x4*)(ap + k0 + 8);        // K = k0 + hh*16 + 8..15
        bf0.q[0] = *(const u32x4*)(bp0 + k0);           // K = k0 + hh*16 + 0..7
        bf0.q[1] = *(const u32x4*)(bp0 + k0 + 8);       // K = k0 + hh*16 + 8..15
        bf1.q[0] = *(const u32x4*)(bp1 + k0);
        bf1.q[1] = *(const u32x4*)(bp1 + k0 + 8);
        acc0 = __builtin_amdgcn_wmma_f32_16x16x32_bf16(false, af.v, false, bf0.v,
                                                       (short)0, acc0, false, false);
        acc1 = __builtin_amdgcn_wmma_f32_16x16x32_bf16(false, af.v, false, bf1.v,
                                                       (short)0, acc1, false, false);
    }

    const int n0 = col0 + nl, n1 = col0 + 16 + nl;
#pragma unroll
    for (int r = 0; r < 8; ++r) {
        int m = row0 + r + 8 * hh;
        if (m >= M) continue;
        float v0 = acc0[r], v1 = acc1[r];
        if (bias)   { v0 += bias[n0]; v1 += bias[n1]; }
        if (extrab) { int bb = m / extraRows;
                      v0 += extrab[(long long)bb * N + n0];
                      v1 += extrab[(long long)bb * N + n1]; }
        if (resid)  { v0 += resid[(long long)m * ldc + n0];
                      v1 += resid[(long long)m * ldc + n1]; }
        v0 = apply_act(v0, act); v1 = apply_act(v1, act);
        if (outF) { outF[(long long)m * ldc + n0] = v0;
                    outF[(long long)m * ldc + n1] = v1; }
        if (outB) { outB[(long long)m * ldc + n0] = (bf16)v0;
                    outB[(long long)m * ldc + n1] = (bf16)v1; }
    }
}

// ---------------------------------------------------------------------------
// Flash attention over 2049 tokens, 8 heads of 64, bf16 QKV, fp32 online
// softmax, WMMA for S = Q K^T and O += P V. 8 waves * 16 q-rows per block.
// qkv layout: [b, t, 1536] with q @ h*64, k @ 512+h*64, v @ 1024+h*64.
// vt layout:  [b, h, d(64), TPAD] (V transposed, padded with zeros).
// ---------------------------------------------------------------------------
__global__ __launch_bounds__(256)
void flash_attn_kernel(const bf16* __restrict__ qkv, const bf16* __restrict__ vt,
                       bf16* __restrict__ obf)
{
    const int b = blockIdx.z, h = blockIdx.y;
    const int wid  = threadIdx.x >> 5;
    const int lane = threadIdx.x & 31;
    const int hh = lane >> 4, nl = lane & 15;
    const int qbase = blockIdx.x * 128 + wid * 16;

    __shared__ bf16 pbuf[8][16 * 32];

    // Q fragments (features 0..31 and 32..63), rows clamped, stores guarded.
    int qtok = qbase + nl; if (qtok > TOK - 1) qtok = TOK - 1;
    const bf16* qrow = qkv + ((long long)(b * TOK + qtok)) * 1536 + h * 64;
    Frag qf0, qf1;
    qf0.q[0] = *(const u32x4*)(qrow + hh * 16);
    qf0.q[1] = *(const u32x4*)(qrow + hh * 16 + 8);
    qf1.q[0] = *(const u32x4*)(qrow + 32 + hh * 16);
    qf1.q[1] = *(const u32x4*)(qrow + 32 + hh * 16 + 8);

    v8f O[4]; O[0] = zero8(); O[1] = zero8(); O[2] = zero8(); O[3] = zero8();
    float mrow[8], lrow[8];
#pragma unroll
    for (int r = 0; r < 8; ++r) { mrow[r] = -1e30f; lrow[r] = 0.f; }

    const bf16* kbase = qkv + (long long)b * TOK * 1536 + 512 + h * 64;
    const bf16* vbase = vt + ((long long)(b * 8 + h)) * 64 * TPAD;

    const int KT = TPAD / 32;           // 65 key tiles of 32
    for (int it = 0; it < KT; ++it) {
        const int kt0 = it * 32;
        int k0tok = kt0 + nl;       if (k0tok > TOK - 1) k0tok = TOK - 1;
        int k1tok = kt0 + 16 + nl;  if (k1tok > TOK - 1) k1tok = TOK - 1;
        const bf16* kr0 = kbase + (long long)k0tok * 1536;
        const bf16* kr1 = kbase + (long long)k1tok * 1536;

        Frag kf;
        v8f S0 = zero8(), S1 = zero8();
        kf.q[0] = *(const u32x4*)(kr0 + hh * 16);
        kf.q[1] = *(const u32x4*)(kr0 + hh * 16 + 8);
        S0 = __builtin_amdgcn_wmma_f32_16x16x32_bf16(false, qf0.v, false, kf.v,
                                                     (short)0, S0, false, false);
        kf.q[0] = *(const u32x4*)(kr0 + 32 + hh * 16);
        kf.q[1] = *(const u32x4*)(kr0 + 32 + hh * 16 + 8);
        S0 = __builtin_amdgcn_wmma_f32_16x16x32_bf16(false, qf1.v, false, kf.v,
                                                     (short)0, S0, false, false);
        kf.q[0] = *(const u32x4*)(kr1 + hh * 16);
        kf.q[1] = *(const u32x4*)(kr1 + hh * 16 + 8);
        S1 = __builtin_amdgcn_wmma_f32_16x16x32_bf16(false, qf0.v, false, kf.v,
                                                     (short)0, S1, false, false);
        kf.q[0] = *(const u32x4*)(kr1 + 32 + hh * 16);
        kf.q[1] = *(const u32x4*)(kr1 + 32 + hh * 16 + 8);
        S1 = __builtin_amdgcn_wmma_f32_16x16x32_bf16(false, qf1.v, false, kf.v,
                                                     (short)0, S1, false, false);

        const bool inv0 = (kt0 + nl)      >= TOK;
        const bool inv1 = (kt0 + 16 + nl) >= TOK;

        float p0[8], p1[8];
#pragma unroll
        for (int r = 0; r < 8; ++r) {
            float s0 = inv0 ? -1e30f : S0[r] * 0.125f;
            float s1 = inv1 ? -1e30f : S1[r] * 0.125f;
            float mx = fmaxf(s0, s1);
#pragma unroll
            for (int sh = 8; sh >= 1; sh >>= 1) mx = fmaxf(mx, __shfl_xor(mx, sh));
            float mnew = fmaxf(mrow[r], mx);
            float e0 = (s0 <= -1e29f) ? 0.f : expf(s0 - mnew);
            float e1 = (s1 <= -1e29f) ? 0.f : expf(s1 - mnew);
            float rs = e0 + e1;
#pragma unroll
            for (int sh = 8; sh >= 1; sh >>= 1) rs += __shfl_xor(rs, sh);
            float alpha = expf(mrow[r] - mnew);
            lrow[r] = lrow[r] * alpha + rs;
            mrow[r] = mnew;
            O[0][r] *= alpha; O[1][r] *= alpha; O[2][r] *= alpha; O[3][r] *= alpha;
            p0[r] = e0; p1[r] = e1;
        }

        __syncthreads();                 // previous reads done before overwrite
#pragma unroll
        for (int r = 0; r < 8; ++r) {
            int rr = r + 8 * hh;
            pbuf[wid][rr * 32 + nl]      = (bf16)p0[r];
            pbuf[wid][rr * 32 + 16 + nl] = (bf16)p1[r];
        }
        __syncthreads();

        Frag pf;
        pf.q[0] = *(const u32x4*)(&pbuf[wid][nl * 32 + hh * 16]);
        pf.q[1] = *(const u32x4*)(&pbuf[wid][nl * 32 + hh * 16 + 8]);

#pragma unroll
        for (int dt = 0; dt < 4; ++dt) {
            const bf16* vp = vbase + (long long)(dt * 16 + nl) * TPAD + kt0 + hh * 16;
            Frag vf;
            vf.q[0] = *(const u32x4*)(vp);
            vf.q[1] = *(const u32x4*)(vp + 8);
            O[dt] = __builtin_amdgcn_wmma_f32_16x16x32_bf16(false, pf.v, false, vf.v,
                                                            (short)0, O[dt], false, false);
        }
    }

#pragma unroll
    for (int r = 0; r < 8; ++r) {
        int tok = qbase + r + 8 * hh;
        if (tok < TOK) {
            float inv = 1.f / lrow[r];
            long long o = ((long long)(b * TOK + tok)) * 512 + h * 64;
            obf[o + 0 * 16 + nl] = (bf16)(O[0][r] * inv);
            obf[o + 1 * 16 + nl] = (bf16)(O[1][r] * inv);
            obf[o + 2 * 16 + nl] = (bf16)(O[2][r] * inv);
            obf[o + 3 * 16 + nl] = (bf16)(O[3][r] * inv);
        }
    }
}

// ---------------------------------------------------------------------------
// LayerNorm over D=512 (one block/row, 256 threads, 2 elems/thread).
// Optional pre-add (residual), post-add (type embedding), fp32 and bf16 outs.
// fp32 out offset: gap>0 -> (row + row/gap + 1)*512 (insert slide token gaps),
// else row*outStride.
// ---------------------------------------------------------------------------
__global__ __launch_bounds__(256)
void ln512_kernel(const float* __restrict__ in, const float* __restrict__ pre,
                  const float* __restrict__ g, const float* __restrict__ be,
                  const float* __restrict__ post,
                  float* outF, long long outStride, int gap,
                  bf16* outB, int R)
{
    const long long row = blockIdx.x;
    const int t = threadIdx.x;
    __shared__ float red[256];
    const float* ip = in + row * 512;
    float v0 = ip[t], v1 = ip[t + 256];
    if (pre) { const float* pp = pre + row * 512; v0 += pp[t]; v1 += pp[t + 256]; }
    red[t] = v0 + v1; __syncthreads();
    for (int o = 128; o > 0; o >>= 1) { if (t < o) red[t] += red[t + o]; __syncthreads(); }
    float mean = red[0] * (1.f / 512.f); __syncthreads();
    float d0 = v0 - mean, d1 = v1 - mean;
    red[t] = d0 * d0 + d1 * d1; __syncthreads();
    for (int o = 128; o > 0; o >>= 1) { if (t < o) red[t] += red[t + o]; __syncthreads(); }
    float rstd = rsqrtf(red[0] * (1.f / 512.f) + 1e-5f);
    float y0 = d0 * rstd * g[t] + be[t];
    float y1 = d1 * rstd * g[t + 256] + be[t + 256];
    if (post) { y0 += post[t]; y1 += post[t + 256]; }
    if (outF) {
        long long o = (gap > 0) ? (row + row / gap + 1) * 512 : row * outStride;
        outF[o + t] = y0; outF[o + t + 256] = y1;
    }
    if (outB) { outB[row * 512 + t] = (bf16)y0; outB[row * 512 + t + 256] = (bf16)y1; }
    (void)R;
}

// ---------------------------------------------------------------------------
// Small dense layers (R<=2 rows): one thread per output element.
// ---------------------------------------------------------------------------
__global__ void small_linear_kernel(const float* in, int ldin,
                                    const float* W, int ldw, const float* bias,
                                    float* out, int ldo, int R, int O, int K, int act)
{
    int i = blockIdx.x * blockDim.x + threadIdx.x;
    if (i >= R * O) return;
    int r = i / O, o = i % O;
    float acc = bias ? bias[o] : 0.f;
    const float* ip = in + (long long)r * ldin;
    const float* wp = W + (long long)o * ldw;
    for (int k = 0; k < K; ++k) acc += ip[k] * wp[k];
    out[(long long)r * ldo + o] = apply_act(acc, act);
}

__global__ void cvt_f32_bf16_kernel(const float* in, bf16* out, long long n) {
    long long i = (long long)blockIdx.x * 256 + threadIdx.x;
    if (i < n) out[i] = (bf16)in[i];
}

// rank-selection top-k (matches lax.top_k: descending, ties by lower index)
__global__ void topk_kernel(const float* scores, int* idx) {
    int b = blockIdx.x;
    __shared__ float sc[2048];
    for (int i = threadIdx.x; i < 2048; i += 256) sc[i] = scores[b * 2048 + i];
    __syncthreads();
    for (int i = threadIdx.x; i < 2048; i += 256) {
        float v = sc[i];
        int rank = 0;
        for (int j = 0; j < 2048; ++j) {
            float u = sc[j];
            rank += (u > v) || (u == v && j < i);
        }
        if (rank < KTOP) idx[b * KTOP + rank] = i;
    }
}

__global__ void gather_topk_kernel(const bf16* p_bf, const int* idx, bf16* ptk) {
    long long i = (long long)blockIdx.x * 256 + threadIdx.x;
    if (i >= 2LL * KTOP * 512) return;
    int c = (int)(i & 511);
    long long r = i >> 9;
    int b = (int)(r / KTOP), j = (int)(r % KTOP);
    int n = idx[b * KTOP + j];
    ptk[i] = p_bf[((long long)(b * 2048 + n) << 9) + c];
}

__global__ void topk_mean_kernel(const float* x, const int* idx, float* outm) {
    int i = blockIdx.x * 256 + threadIdx.x;
    if (i >= 1024) return;
    int b = i >> 9, d = i & 511;
    float s = 0.f;
    for (int j = 0; j < KTOP; ++j) {
        int n = idx[b * KTOP + j];
        s += x[((long long)(b * TOK + 1 + n)) * 512 + d];
    }
    outm[i] = s * (1.f / KTOP);
}

// cross attention with a single query token per (b,h)
__global__ void cross_attn_kernel(const float* qvec, const float* kh,
                                  const float* vh, float* co)
{
    int b = blockIdx.x, h = blockIdx.y, t = threadIdx.x;
    __shared__ float q[64];
    __shared__ float lg[KTOP];
    __shared__ float red[256];
    if (t < 64) q[t] = qvec[b * 512 + h * 64 + t];
    __syncthreads();
    float lmax = -1e30f;
    for (int j = t; j < KTOP; j += 256) {
        const float* kr = kh + ((long long)(b * KTOP + j)) * 512 + h * 64;
        float d = 0.f;
        for (int c = 0; c < 64; ++c) d += q[c] * kr[c];
        d *= 0.125f;
        lg[j] = d; lmax = fmaxf(lmax, d);
    }
    red[t] = lmax; __syncthreads();
    for (int o = 128; o > 0; o >>= 1) { if (t < o) red[t] = fmaxf(red[t], red[t + o]); __syncthreads(); }
    float m = red[0]; __syncthreads();
    float lsum = 0.f;
    for (int j = t; j < KTOP; j += 256) { float e = expf(lg[j] - m); lg[j] = e; lsum += e; }
    red[t] = lsum; __syncthreads();
    for (int o = 128; o > 0; o >>= 1) { if (t < o) red[t] += red[t + o]; __syncthreads(); }
    float Z = red[0]; __syncthreads();
    if (t < 64) {
        float acc = 0.f;
        for (int j = 0; j < KTOP; ++j)
            acc += lg[j] * vh[((long long)(b * KTOP + j)) * 512 + h * 64 + t];
        co[b * 512 + h * 64 + t] = acc / Z;
    }
}

// V transpose + zero-pad into [b,h,64,TPAD]
__global__ void vtrans_kernel(const bf16* qkv, bf16* vt) {
    long long i = (long long)blockIdx.x * 256 + threadIdx.x;
    const long long TOT = 2LL * 8 * 64 * TPAD;
    if (i >= TOT) return;
    int tp = (int)(i % TPAD);
    long long r = i / TPAD;
    int d = (int)(r % 64); r /= 64;
    int h = (int)(r % 8);
    int b = (int)(r / 8);
    bf16 v = (bf16)0.f;
    if (tp < TOK) v = qkv[((long long)(b * TOK + tp)) * 1536 + 1024 + h * 64 + d];
    vt[i] = v;
}

__global__ void rowdot_kernel(const float* A, const float* B2, const float* w,
                              const float* bias, float* out, int R, int Kd)
{
    int r = blockIdx.x * blockDim.x + threadIdx.x;
    if (r >= R) return;
    const float* a = A + (long long)r * Kd;
    const float* b2 = B2 ? B2 + (long long)r * Kd : nullptr;
    float acc = bias ? bias[0] : 0.f;
    for (int c = 0; c < Kd; ++c) acc += a[c] * (b2 ? b2[c] : 1.f) * w[c];
    out[r] = acc;
}

__global__ void softmax2048_kernel(float* s) {
    int b = blockIdx.x, t = threadIdx.x;
    __shared__ float red[256];
    float* row = s + b * 2048;
    float mx = -1e30f;
    for (int i = t; i < 2048; i += 256) mx = fmaxf(mx, row[i]);
    red[t] = mx; __syncthreads();
    for (int o = 128; o > 0; o >>= 1) { if (t < o) red[t] = fmaxf(red[t], red[t + o]); __syncthreads(); }
    float m = red[0]; __syncthreads();
    float sm = 0.f;
    for (int i = t; i < 2048; i += 256) { float e = expf(row[i] - m); row[i] = e; sm += e; }
    red[t] = sm; __syncthreads();
    for (int o = 128; o > 0; o >>= 1) { if (t < o) red[t] += red[t + o]; __syncthreads(); }
    float inv = 1.f / red[0]; __syncthreads();
    for (int i = t; i < 2048; i += 256) row[i] *= inv;
}

__global__ void mil_pool_kernel(const float* probs, const float* x, float* out) {
    int i = blockIdx.x * 256 + threadIdx.x;
    if (i >= 1024) return;
    int b = i >> 9, d = i & 511;
    float s = 0.f;
    for (int n = 0; n < 2048; ++n)
        s += probs[b * 2048 + n] * x[((long long)(b * TOK + 1 + n)) * 512 + d];
    out[i] = s;
}

__global__ void extract_pbf_kernel(const float* x, bf16* out) {
    long long i = (long long)blockIdx.x * 256 + threadIdx.x;
    if (i >= 2LL * 2048 * 512) return;
    long long r = i >> 9; int c = (int)(i & 511);
    out[i] = (bf16)x[(r + r / 2048 + 1) * 512 + c];
}

__global__ void copy_pfinal_kernel(const float* x, float* out) {
    long long i = (long long)blockIdx.x * 256 + threadIdx.x;
    if (i >= 2LL * 2048 * 512) return;
    long long r = i >> 9; int c = (int)(i & 511);
    out[i] = x[(r + r / 2048 + 1) * 512 + c];
}

__global__ void build_multi_kernel(const float* x, const float* ptkm,
                                   const float* pooled, float* multi)
{
    int i = blockIdx.x * 256 + threadIdx.x;
    if (i >= 2 * 1536) return;
    int b = i / 1536, c = i % 1536;
    float v;
    if (c < 512)       v = x[(long long)b * TOK * 512 + c];
    else if (c < 1024) v = ptkm[b * 512 + (c - 512)];
    else               v = pooled[b * 512 + (c - 1024)];
    multi[i] = v;
}

// ---------------------------------------------------------------------------
// Host orchestration
// ---------------------------------------------------------------------------
static inline void gemmw(hipStream_t st, const bf16* A, int lda, const bf16* W, int ldw,
                         const float* bias, const float* extrab, int extraRows,
                         const float* resid, float* outF, bf16* outB, int ldc,
                         int M, int N, int K, int act)
{
    dim3 grid((M + 63) / 64, N / 64);
    gemm_wmma_kernel<<<grid, 256, 0, st>>>(A, lda, W, ldw, bias, extrab, extraRows,
                                           resid, outF, outB, ldc, M, N, K, act);
}

static inline void cvtw(hipStream_t st, const float* src, bf16* dst, long long n) {
    cvt_f32_bf16_kernel<<<(unsigned)((n + 255) / 256), 256, 0, st>>>(src, dst, n);
}

extern "C" void kernel_launch(void* const* d_in, const int* in_sizes, int n_in,
                              void* d_out, int out_size, void* d_ws, size_t ws_size,
                              hipStream_t stream)
{
    // ---- input leaf mapping (setup_inputs dict order, params depth-first) ----
    // 0 slide_feat(2,768) 1 patch_feats(2,2048,512)
    // 2-5 proj_slide{w,b,ln.w,ln.b} 6-9 proj_patch{w,b,ln.w,ln.b}
    // 10 type_slide 11 type_patch 12-13 coarse1{w,b} 14-15 coarse2{w,b}
    // 16-17 cross.in{w,b} 18-19 cross.out{w,b} 20-21 fine_norm{w,b}
    // 22+12*l: layer l {attn_in.w,b, attn_out.w,b, ln1.w,b, ln2.w,b, ff1.w,b, ff2.w,b}
    // 46-51 mil{att_a.w,b, att_b.w,b, gate.w,b} 52-53 fus1 54-55 fus2 56-57 fus_ln
    auto P = [&](int i) -> const float* { return (const float*)d_in[i]; };
    (void)in_sizes; (void)n_in; (void)out_size;

    char* wsb = (char*)d_ws;
    size_t off = 0;
    auto alloc = [&](size_t bytes) -> void* {
        void* p = wsb + off;
        off = (off + bytes + 255) & ~(size_t)255;
        return p;
    };

    // bf16 weight copies
    bf16* wb_pp = (bf16*)alloc(512LL * 512 * 2);
    bf16* wb_c1 = (bf16*)alloc(256LL * 1024 * 2);
    bf16* wb_ci = (bf16*)alloc(1536LL * 512 * 2);
    bf16* wb_ai[2]; bf16* wb_ao[2]; bf16* wb_f1[2]; bf16* wb_f2[2];
    for (int l = 0; l < 2; ++l) {
        wb_ai[l] = (bf16*)alloc(1536LL * 512 * 2);
        wb_ao[l] = (bf16*)alloc(512LL * 512 * 2);
        wb_f1[l] = (bf16*)alloc(2048LL * 512 * 2);
        wb_f2[l] = (bf16*)alloc(512LL * 2048 * 2);
    }
    bf16* wb_ma = (bf16*)alloc(256LL * 512 * 2);
    bf16* wb_mg = (bf16*)alloc(256LL * 512 * 2);

    // activations
    bf16*  pf_bf  = (bf16*)alloc(4096LL * 512 * 2);
    bf16*  p_bf   = (bf16*)alloc(4096LL * 512 * 2);
    float* x      = (float*)alloc(2LL * TOK * 512 * 4);
    float* tmpF   = (float*)alloc(4096LL * 512 * 4);
    float* slin   = (float*)alloc(1024 * 4);
    float* sbuf   = (float*)alloc(1024 * 4);
    float* csb    = (float*)alloc(512 * 4);
    float* scores = (float*)alloc(4096 * 4);
    int*   idx    = (int*)  alloc(2LL * KTOP * 4);
    bf16*  ptk_bf = (bf16*)alloc(2LL * KTOP * 512 * 2);
    float* ptkm   = (float*)alloc(1024 * 4);
    float* qvec   = (float*)alloc(1024 * 4);
    float* co     = (float*)alloc(1024 * 4);
    float* fo     = (float*)alloc(1024 * 4);
    float* kh     = (float*)alloc(2LL * KTOP * 512 * 4);
    float* vh     = (float*)alloc(2LL * KTOP * 512 * 4);
    bf16*  y_bf   = (bf16*)alloc(2LL * TOK * 512 * 2);
    bf16*  qkv_bf = (bf16*)alloc(2LL * TOK * 1536 * 2);
    bf16*  vt     = (bf16*)alloc(2LL * 8 * 64 * TPAD * 2);
    bf16*  o_bf   = (bf16*)alloc(2LL * TOK * 512 * 2);
    bf16*  ffh_bf = (bf16*)alloc(2LL * TOK * 2048 * 2);
    float* pooled = (float*)alloc(1024 * 4);
    float* multi  = (float*)alloc(2LL * 1536 * 4);
    float* fm1    = (float*)alloc(2LL * 1024 * 4);
    float* fm2    = (float*)alloc(1024 * 4);
    (void)ws_size;

    // ---- weight + input conversions ----
    cvtw(stream, P(6),  wb_pp, 512LL * 512);
    cvtw(stream, P(12), wb_c1, 256LL * 1024);
    cvtw(stream, P(16), wb_ci, 1536LL * 512);
    for (int l = 0; l < 2; ++l) {
        int pb = 22 + 12 * l;
        cvtw(stream, P(pb + 0),  wb_ai[l], 1536LL * 512);
        cvtw(stream, P(pb + 2),  wb_ao[l], 512LL * 512);
        cvtw(stream, P(pb + 8),  wb_f1[l], 2048LL * 512);
        cvtw(stream, P(pb + 10), wb_f2[l], 512LL * 2048);
    }
    cvtw(stream, P(46), wb_ma, 256LL * 512);
    cvtw(stream, P(50), wb_mg, 256LL * 512);
    cvtw(stream, P(1),  pf_bf, 4096LL * 512);

    // ---- slide projection: linear -> LN -> +type_slide ----
    small_linear_kernel<<<(2 * 512 + 255) / 256, 256, 0, stream>>>(
        P(0), 768, P(2), 768, P(3), slin, 512, 2, 512, 768, ACT_NONE);
    ln512_kernel<<<2, 256, 0, stream>>>(slin, nullptr, P(4), P(5), P(10),
                                        sbuf, 512, 0, nullptr, 2);

    // ---- patch projection: WMMA GEMM -> LN -> +type_patch ----
    gemmw(stream, pf_bf, 512, wb_pp, 512, P(7), nullptr, 0, nullptr,
          tmpF, nullptr, 512, 4096, 512, 512, ACT_NONE);
    // fp32 into x rows [b,1+n]; bf16 copy into p_bf
    ln512_kernel<<<4096, 256, 0, stream>>>(tmpF, nullptr, P(8), P(9), P(11),
                                           x, 0, 2048, p_bf, 4096);

    // ---- coarse scorer (fold s-half of concat into per-batch bias) ----
    small_linear_kernel<<<(2 * 256 + 255) / 256, 256, 0, stream>>>(
        sbuf, 512, P(12) + 512, 1024, P(13), csb, 256, 2, 256, 512, ACT_NONE);
    gemmw(stream, p_bf, 512, wb_c1, 1024, nullptr, csb, 2048, nullptr,
          tmpF, nullptr, 256, 4096, 256, 512, ACT_GELU);
    rowdot_kernel<<<16, 256, 0, stream>>>(tmpF, nullptr, P(14), P(15), scores, 4096, 256);

    // ---- top-k + gather + mean ----
    topk_kernel<<<2, 256, 0, stream>>>(scores, idx);
    gather_topk_kernel<<<(unsigned)((2LL * KTOP * 512 + 255) / 256), 256, 0, stream>>>(
        p_bf, idx, ptk_bf);
    topk_mean_kernel<<<4, 256, 0, stream>>>(x, idx, ptkm);

    // ---- cross attention (q = slide token) ----
    small_linear_kernel<<<(2 * 512 + 255) / 256, 256, 0, stream>>>(
        sbuf, 512, P(16), 512, P(17), qvec, 512, 2, 512, 512, ACT_NONE);
    gemmw(stream, ptk_bf, 512, wb_ci + 512LL * 512, 512, P(17) + 512, nullptr, 0,
          nullptr, kh, nullptr, 512, 2 * KTOP, 512, 512, ACT_NONE);
    gemmw(stream, ptk_bf, 512, wb_ci + 1024LL * 512, 512, P(17) + 1024, nullptr, 0,
          nullptr, vh, nullptr, 512, 2 * KTOP, 512, 512, ACT_NONE);
    cross_attn_kernel<<<dim3(2, 8), 256, 0, stream>>>(qvec, kh, vh, co);
    small_linear_kernel<<<(2 * 512 + 255) / 256, 256, 0, stream>>>(
        co, 512, P(18), 512, P(19), fo, 512, 2, 512, 512, ACT_NONE);
    // s_ref = LN(s + fine_out) -> x[b,0,:]
    ln512_kernel<<<2, 256, 0, stream>>>(sbuf, fo, P(20), P(21), nullptr,
                                        x, (long long)TOK * 512, 0, nullptr, 2);

    // ---- 2 transformer encoder layers over 2049 tokens ----
    const int MT = 2 * TOK;  // 4098 rows
    for (int l = 0; l < 2; ++l) {
        int pb = 22 + 12 * l;
        // y = LN1(x) (bf16)
        ln512_kernel<<<MT, 256, 0, stream>>>(x, nullptr, P(pb + 4), P(pb + 5), nullptr,
                                             nullptr, 0, 0, y_bf, MT);
        // qkv
        gemmw(stream, y_bf, 512, wb_ai[l], 512, P(pb + 1), nullptr, 0, nullptr,
              nullptr, qkv_bf, 1536, MT, 1536, 512, ACT_NONE);
        // V transpose for WMMA B-fragments
        vtrans_kernel<<<(unsigned)((2LL * 8 * 64 * TPAD + 255) / 256), 256, 0, stream>>>(
            qkv_bf, vt);
        // flash attention
        flash_attn_kernel<<<dim3((TOK + 127) / 128, 8, 2), 256, 0, stream>>>(
            qkv_bf, vt, o_bf);
        // out proj + residual (x += ...)
        gemmw(stream, o_bf, 512, wb_ao[l], 512, P(pb + 3), nullptr, 0, x,
              x, nullptr, 512, MT, 512, 512, ACT_NONE);
        // y = LN2(x)
        ln512_kernel<<<MT, 256, 0, stream>>>(x, nullptr, P(pb + 6), P(pb + 7), nullptr,
                                             nullptr, 0, 0, y_bf, MT);
        // FF1 (relu) -> FF2 (+residual)
        gemmw(stream, y_bf, 512, wb_f1[l], 512, P(pb + 9), nullptr, 0, nullptr,
              nullptr, ffh_bf, 2048, MT, 2048, 512, ACT_RELU);
        gemmw(stream, ffh_bf, 2048, wb_f2[l], 2048, P(pb + 11), nullptr, 0, x,
              x, nullptr, 512, MT, 512, 2048, ACT_NONE);
    }

    // ---- MIL gated attention pooling over p_final ----
    extract_pbf_kernel<<<(unsigned)((2LL * 2048 * 512 + 255) / 256), 256, 0, stream>>>(
        x, p_bf);
    gemmw(stream, p_bf, 512, wb_ma, 512, P(47), nullptr, 0, nullptr,
          tmpF, nullptr, 256, 4096, 256, 512, ACT_TANH);
    gemmw(stream, p_bf, 512, wb_mg, 512, P(51), nullptr, 0, nullptr,
          tmpF + 4096LL * 256, nullptr, 256, 4096, 256, 512, ACT_SIGM);
    rowdot_kernel<<<16, 256, 0, stream>>>(tmpF, tmpF + 4096LL * 256, P(48), P(49),
                                          scores, 4096, 256);
    softmax2048_kernel<<<2, 256, 0, stream>>>(scores);
    mil_pool_kernel<<<4, 256, 0, stream>>>(scores, x, pooled);

    // ---- fusion head ----
    build_multi_kernel<<<(2 * 1536 + 255) / 256, 256, 0, stream>>>(x, ptkm, pooled, multi);
    small_linear_kernel<<<(2 * 1024 + 255) / 256, 256, 0, stream>>>(
        multi, 1536, P(52), 1536, P(53), fm1, 1024, 2, 1024, 1536, ACT_GELU);
    small_linear_kernel<<<(2 * 512 + 255) / 256, 256, 0, stream>>>(
        fm1, 1024, P(54), 1024, P(55), fm2, 512, 2, 512, 1024, ACT_NONE);
    ln512_kernel<<<2, 256, 0, stream>>>(fm2, nullptr, P(56), P(57), nullptr,
                                        (float*)d_out, 512, 0, nullptr, 2);
    // p_final
    copy_pfinal_kernel<<<(unsigned)((2LL * 2048 * 512 + 255) / 256), 256, 0, stream>>>(
        x, (float*)d_out + 1024);
}
